// CellEncoder_1864015806865
// MI455X (gfx1250) — matrix operand order repaired and checked by reading, not compiled, not run on hardware
//
#include <hip/hip_runtime.h>
#include <hip/hip_bf16.h>

#define NEG_SLOPE 0.2f

typedef __attribute__((ext_vector_type(16))) _Float16 v16h;
typedef __attribute__((ext_vector_type(8)))  float    v8f;

// ---- monotonic float<->uint mapping so float max can use native u32 atomic max ----
__device__ __forceinline__ unsigned f32_to_ord(float f) {
  unsigned u = __float_as_uint(f);
  return (u & 0x80000000u) ? ~u : (u | 0x80000000u);
}
__device__ __forceinline__ float ord_to_f32(unsigned o) {
  unsigned u = (o & 0x80000000u) ? (o & 0x7FFFFFFFu) : ~o;
  return __uint_as_float(u);
}

// ---- native fire-and-forget f32 atomic add (avoid CAS-loop lowering) ----
__device__ __forceinline__ void gatomic_add_f32(float* p, float v) {
  asm volatile("global_atomic_add_f32 %0, %1, off" :: "v"(p), "v"(v) : "memory");
}

// ============================================================================
// GEMM: hs[N,32] = h[N,DIN] @ W[DIN,32], f16 WMMA with f32 accumulate.
// One wave per 16-row tile; two v_wmma_f32_16x16x32_f16 per tile (N-halves).
// DIN is a template parameter (3 or 32) so all fragment loads are branchless.
// ============================================================================
template <int DIN>
__global__ void gat_gemm_wmma(const float* __restrict__ h,
                              const float* __restrict__ Wl,
                              float* __restrict__ hs, int ntiles) {
  int wave = threadIdx.x >> 5;
  int tile = blockIdx.x * 4 + wave;
  if (tile >= ntiles) return;          // wave-uniform exit; EXEC all-ones at WMMA
  int lane = threadIdx.x & 31;
  int hsel = lane >> 4;                // which K-half this lane carries
  int m    = lane & 15;                // row (A) / column (B,C,D) index

  // ---- A fragment (16x32 f16): lane holds row M = lane%16.
  // lanes 0-15: K in {0..7, 16..23}; lanes 16-31: K in {8..15, 24..31}.
  v16h a;
  if (DIN == 32) {
    const float4* h4 = (const float4*)(h + (size_t)(tile * 16 + m) * 32);
    float4 q0 = h4[hsel * 2 + 0];      // K = 8*hsel + 0..3
    float4 q1 = h4[hsel * 2 + 1];      // K = 8*hsel + 4..7
    float4 q2 = h4[hsel * 2 + 4];      // K = 16 + 8*hsel + 0..3
    float4 q3 = h4[hsel * 2 + 5];      // K = 16 + 8*hsel + 4..7
    a[0]  = (_Float16)q0.x; a[1]  = (_Float16)q0.y; a[2]  = (_Float16)q0.z; a[3]  = (_Float16)q0.w;
    a[4]  = (_Float16)q1.x; a[5]  = (_Float16)q1.y; a[6]  = (_Float16)q1.z; a[7]  = (_Float16)q1.w;
    a[8]  = (_Float16)q2.x; a[9]  = (_Float16)q2.y; a[10] = (_Float16)q2.z; a[11] = (_Float16)q2.w;
    a[12] = (_Float16)q3.x; a[13] = (_Float16)q3.y; a[14] = (_Float16)q3.z; a[15] = (_Float16)q3.w;
  } else {
    // DIN == 3: only K = 0,1,2 exist, and those live in lanes 0-15 (hsel==0).
    const float* hrow = h + (size_t)(tile * 16 + m) * DIN;
    float mask = (hsel == 0) ? 1.0f : 0.0f;
    float h0 = hrow[0] * mask, h1 = hrow[1] * mask, h2 = hrow[2] * mask;
    a[0] = (_Float16)h0; a[1] = (_Float16)h1; a[2] = (_Float16)h2;
#pragma unroll
    for (int j = 3; j < 16; ++j) a[j] = (_Float16)0.0f;
  }

  // ---- B fragments (32x16 f16 each): lane%16 = column; K order matches A (8*hsel + j, then 16 + 8*hsel + j-8).
  v16h b0, b1;
  if (DIN == 32) {
#pragma unroll
    for (int j = 0; j < 16; ++j) {
      int K = (j < 8) ? (8 * hsel + j) : (16 + 8 * hsel + (j - 8));
      b0[j] = (_Float16)Wl[K * 32 + m];
      b1[j] = (_Float16)Wl[K * 32 + 16 + m];
    }
  } else {
    float mask = (hsel == 0) ? 1.0f : 0.0f;   // K = 8*hsel + j (j<8) < 3 only if hsel==0, j<3
#pragma unroll
    for (int j = 0; j < 16; ++j) {
      float w0 = (j < DIN) ? Wl[j * 32 + m] * mask      : 0.0f;
      float w1 = (j < DIN) ? Wl[j * 32 + 16 + m] * mask : 0.0f;
      b0[j] = (_Float16)w0;
      b1[j] = (_Float16)w1;
    }
  }

  v8f c0 = {}, c1 = {};
  c0 = __builtin_amdgcn_wmma_f32_16x16x32_f16(false, a, false, b0, (short)0, c0, false, false);
  c1 = __builtin_amdgcn_wmma_f32_16x16x32_f16(false, a, false, b1, (short)0, c1, false, false);

  // C/D layout: VGPR r -> M = r + 8*(lane/16), N = lane%16.
  float* obase = hs + (size_t)(tile * 16 + 8 * hsel) * 32 + m;
#pragma unroll
  for (int r = 0; r < 8; ++r) {
    obase[(size_t)r * 32]      = c0[r];
    obase[(size_t)r * 32 + 16] = c1[r];
  }
}

// alpha_src[i] = hs[i,:] . a_s ; alpha_dst[i] = hs[i,:] . a_d
__global__ void alpha_kernel(const float* __restrict__ hs,
                             const float* __restrict__ a_s,
                             const float* __restrict__ a_d,
                             float* __restrict__ as_o, float* __restrict__ ad_o, int n) {
  int i = blockIdx.x * blockDim.x + threadIdx.x;
  if (i >= n) return;
  const float4* row = (const float4*)(hs + (size_t)i * 32);
  float s0 = 0.0f, s1 = 0.0f;
#pragma unroll
  for (int q = 0; q < 8; ++q) {
    float4 v = row[q];
    s0 += v.x * a_s[q * 4 + 0] + v.y * a_s[q * 4 + 1] + v.z * a_s[q * 4 + 2] + v.w * a_s[q * 4 + 3];
    s1 += v.x * a_d[q * 4 + 0] + v.y * a_d[q * 4 + 1] + v.z * a_d[q * 4 + 2] + v.w * a_d[q * 4 + 3];
  }
  as_o[i] = s0;
  ad_o[i] = s1;
}

// zero the output accumulator (n*32 floats) and segment arrays (n each)
__global__ void zero_kernel(float* __restrict__ outb, unsigned* __restrict__ segmax,
                            float* __restrict__ segsum, int n) {
  int i = blockIdx.x * blockDim.x + threadIdx.x;
  if (i < n * 32) outb[i] = 0.0f;
  if (i < n) { segmax[i] = 0u; segsum[i] = 0.0f; }
}

__device__ __forceinline__ void edge_sd(const int* esrc, const int* edst, int i, int E,
                                        int& s, int& d) {
  if (i < E) { s = esrc[i]; d = edst[i]; }
  else       { s = i - E;   d = i - E;   }   // self loops appended
}

__device__ __forceinline__ float edge_logit(const float* as_, const float* ad_, int s, int d) {
  float e = as_[s] + ad_[d];
  return (e > 0.0f) ? e : NEG_SLOPE * e;
}

// pass 1: segment max of leaky-relu logits over dst
__global__ void edge_pass1(const int* __restrict__ esrc, const int* __restrict__ edst,
                           int E, int EE,
                           const float* __restrict__ as_, const float* __restrict__ ad_,
                           unsigned* __restrict__ segmax) {
  int i = blockIdx.x * blockDim.x + threadIdx.x;
  if (i >= EE) return;
  int s, d; edge_sd(esrc, edst, i, E, s, d);
  float e = edge_logit(as_, ad_, s, d);
  atomicMax(&segmax[d], f32_to_ord(e));     // native global_atomic_max_u32
}

// pass 2: segment sum of exp(e - max)
__global__ void edge_pass2(const int* __restrict__ esrc, const int* __restrict__ edst,
                           int E, int EE,
                           const float* __restrict__ as_, const float* __restrict__ ad_,
                           const unsigned* __restrict__ segmax, float* __restrict__ segsum) {
  int i = blockIdx.x * blockDim.x + threadIdx.x;
  if (i >= EE) return;
  int s, d; edge_sd(esrc, edst, i, E, s, d);
  float e = edge_logit(as_, ad_, s, d);
  float ex = __expf(e - ord_to_f32(segmax[d]));
  gatomic_add_f32(&segsum[d], ex);
}

// pass 3: out[dst,:] += softmax_weight * hs[src,:]
__global__ void edge_pass3(const int* __restrict__ esrc, const int* __restrict__ edst,
                           int E, int EE,
                           const float* __restrict__ as_, const float* __restrict__ ad_,
                           const unsigned* __restrict__ segmax, const float* __restrict__ segsum,
                           const float* __restrict__ hs, float* __restrict__ outb) {
  int i = blockIdx.x * blockDim.x + threadIdx.x;
  if (i >= EE) return;
  int s, d; edge_sd(esrc, edst, i, E, s, d);
  float e = edge_logit(as_, ad_, s, d);
  float w = __expf(e - ord_to_f32(segmax[d])) / segsum[d];
  const float4* hrow = (const float4*)(hs + (size_t)s * 32);
  float* orow = outb + (size_t)d * 32;
#pragma unroll
  for (int q = 0; q < 8; ++q) {
    float4 v = hrow[q];
    gatomic_add_f32(orow + q * 4 + 0, w * v.x);
    gatomic_add_f32(orow + q * 4 + 1, w * v.y);
    gatomic_add_f32(orow + q * 4 + 2, w * v.z);
    gatomic_add_f32(orow + q * 4 + 3, w * v.w);
  }
}

// out = relu(out + bias) in place
__global__ void finalize_kernel(float* __restrict__ outb, const float* __restrict__ bias, int n32) {
  int i = blockIdx.x * blockDim.x + threadIdx.x;
  if (i >= n32) return;
  float v = outb[i] + bias[i & 31];
  outb[i] = v > 0.0f ? v : 0.0f;
}

extern "C" void kernel_launch(void* const* d_in, const int* in_sizes, int n_in,
                              void* d_out, int out_size, void* d_ws, size_t ws_size,
                              hipStream_t stream) {
  const float* x   = (const float*)d_in[0];
  const int*   ei  = (const int*)d_in[1];
  const float* W0  = (const float*)d_in[2];
  const float* as0 = (const float*)d_in[3];
  const float* ad0 = (const float*)d_in[4];
  const float* b0  = (const float*)d_in[5];
  const float* Wr  = (const float*)d_in[6];
  const float* asr = (const float*)d_in[7];
  const float* adr = (const float*)d_in[8];
  const float* br  = (const float*)d_in[9];
  float* out = (float*)d_out;

  const int N = in_sizes[0] / 3;
  const int E = in_sizes[1] / 2;
  const int D = 32;
  const int EE = E + N;

  // workspace layout (floats): hs[N*32] | hA[N*32] | al_s[N] | al_d[N] | segsum[N] | segmax[N]
  float* ws      = (float*)d_ws;
  float* hs      = ws;
  float* hA      = hs + (size_t)N * D;
  float* al_s    = hA + (size_t)N * D;
  float* al_d    = al_s + N;
  float* segsum  = al_d + N;
  unsigned* segmax = (unsigned*)(segsum + N);

  const int* esrc = ei;
  const int* edst = ei + E;

  const int tiles = (N + 15) / 16;
  dim3 gemmGrid((tiles + 3) / 4), gemmBlk(128);
  const int gN   = (N + 255) / 256;
  const int gN32 = (N * 32 + 255) / 256;
  const int gE   = (EE + 255) / 256;

  for (int l = 0; l < 3; ++l) {
    const float* hin  = (l == 0) ? x   : hA;
    const float* Wl   = (l == 0) ? W0  : (Wr  + (size_t)(l - 1) * D * D);
    const float* a_sl = (l == 0) ? as0 : (asr + (size_t)(l - 1) * D);
    const float* a_dl = (l == 0) ? ad0 : (adr + (size_t)(l - 1) * D);
    const float* bl   = (l == 0) ? b0  : (br  + (size_t)(l - 1) * D);
    float* outb = (l == 2) ? out : hA;   // hin (hA) is dead after the gemm kernel

    if (l == 0)
      gat_gemm_wmma<3><<<gemmGrid, gemmBlk, 0, stream>>>(hin, Wl, hs, tiles);
    else
      gat_gemm_wmma<32><<<gemmGrid, gemmBlk, 0, stream>>>(hin, Wl, hs, tiles);

    alpha_kernel<<<gN, 256, 0, stream>>>(hs, a_sl, a_dl, al_s, al_d, N);
    zero_kernel<<<gN32, 256, 0, stream>>>(outb, segmax, segsum, N);
    edge_pass1<<<gE, 256, 0, stream>>>(esrc, edst, E, EE, al_s, al_d, segmax);
    edge_pass2<<<gE, 256, 0, stream>>>(esrc, edst, E, EE, al_s, al_d, segmax, segsum);
    edge_pass3<<<gE, 256, 0, stream>>>(esrc, edst, E, EE, al_s, al_d, segmax, segsum, hs, outb);
    finalize_kernel<<<gN32, 256, 0, stream>>>(outb, bl, N * 32);
  }
}
